// Attention_26207890440384
// MI455X (gfx1250) — matrix operand order repaired and checked by reading, not compiled, not run on hardware
//
#include <hip/hip_runtime.h>
#include <hip/hip_bf16.h>
#include <math.h>

// ---------------------------------------------------------------------------
// Problem constants (reference: H=8, DQK=128, DV=192, DIM=768, n=512)
// ---------------------------------------------------------------------------
#define H_HEADS 8
#define DQK     128
#define DV      192
#define DIM     768
#define NSEQ    512
#define NQKV    (H_HEADS*DQK + DQK + DV)   // 1344
#define HDV     (H_HEADS*DV)               // 1536

typedef __attribute__((ext_vector_type(16))) _Float16 v16h;
typedef __attribute__((ext_vector_type(8)))  float    v8f;
typedef __attribute__((ext_vector_type(4)))  float    f32x4;   // native vec for NT loads

// ---------------------------------------------------------------------------
// WMMA helpers (gfx1250, wave32). D = A(16x32,f16) * B(32x16,f16) + C(16x16,f32)
// Layouts: K order is contiguous (k = 16*half + s), same in A and B.
// ---------------------------------------------------------------------------
__device__ __forceinline__ v8f wmma16x16x32(v16h a, v16h b, v8f c) {
  return __builtin_amdgcn_wmma_f32_16x16x32_f16(
      /*neg_a=*/false, a, /*neg_b=*/false, b,
      /*c_mod=*/(short)0, c, /*reuse_a=*/false, /*reuse_b=*/false);
}

// A fragment: src is row-major [16][>=32] f16, leading dim ld.
// Lane L: M = L%16, half = L/16; VGPR v: K = half*16 + 2v.
__device__ __forceinline__ v16h frag_a_rowmajor(const _Float16* src, int ld) {
  int lane = threadIdx.x & 31;
  int m = lane & 15, half = lane >> 4;
  v16h a;
#pragma unroll
  for (int v = 0; v < 8; ++v) {
    int k = (half << 4) + (v << 1);
    a[2*v]   = src[m*ld + k];
    a[2*v+1] = src[m*ld + k + 1];
  }
  return a;
}

// B fragment from row-major B[k][n] (k-major storage, leading dim ld).
// Lane L: N = L%16, half = L/16; VGPR v: K = half*16 + 2v.
__device__ __forceinline__ v16h frag_b_rowmajor(const _Float16* src, int ld) {
  int lane = threadIdx.x & 31;
  int n = lane & 15, half = lane >> 4;
  v16h b;
#pragma unroll
  for (int v = 0; v < 8; ++v) {
    int k = (half << 4) + (v << 1);
    b[2*v]   = src[k*ld + n];
    b[2*v+1] = src[(k+1)*ld + n];
  }
  return b;
}

// B fragment where memory holds B^T: element B[k][n] = src[n*ld + k]
// (used for Q*K^T where K is stored [n][d] row-major).
__device__ __forceinline__ v16h frag_b_transposed(const _Float16* src, int ld) {
  int lane = threadIdx.x & 31;
  int n = lane & 15, half = lane >> 4;
  v16h b;
#pragma unroll
  for (int v = 0; v < 8; ++v) {
    int k = (half << 4) + (v << 1);
    b[2*v]   = src[n*ld + k];
    b[2*v+1] = src[n*ld + k + 1];
  }
  return b;
}

// ---------------------------------------------------------------------------
// fp32 -> f16 conversion
// ---------------------------------------------------------------------------
__global__ void cvt_f32_f16(const float* __restrict__ in, _Float16* __restrict__ out, int n) {
  int i = blockIdx.x * blockDim.x + threadIdx.x;
  if (i < n) out[i] = (_Float16)in[i];
}

// ---------------------------------------------------------------------------
// Generic WMMA GEMM: C[M][N] (f32) = A[M][K] * B[K][N], f16 inputs.
// Block = 4 waves, each wave owns one 16x16 tile; block covers 16(M) x 64(N).
// K is always a multiple of 128 here -> unroll 4 for load/WMMA overlap.
// ---------------------------------------------------------------------------
__global__ void __launch_bounds__(128) gemm_f16_wmma(
    const _Float16* __restrict__ A, const _Float16* __restrict__ B,
    float* __restrict__ C, int M, int N, int K) {
  int wave = threadIdx.x >> 5, lane = threadIdx.x & 31;
  int m0 = blockIdx.x * 16;
  int n0 = (blockIdx.y * 4 + wave) * 16;
  if (m0 >= M || n0 >= N) return;   // wave-uniform guard
  v8f acc = {0,0,0,0,0,0,0,0};
#pragma unroll 4
  for (int k0 = 0; k0 < K; k0 += 32) {
    v16h a = frag_a_rowmajor(A + (size_t)m0 * K + k0, K);
    v16h b = frag_b_rowmajor(B + (size_t)k0 * N + n0, N);
    acc = wmma16x16x32(a, b, acc);
  }
  int nn = lane & 15, half = lane >> 4;
#pragma unroll
  for (int r = 0; r < 8; ++r)
    C[(size_t)(m0 + half*8 + r) * N + n0 + nn] = acc[r];
}

// ---------------------------------------------------------------------------
// Per-token LayerNorms + rotary; emits f16 q [h][n][128], k [n][128], v [n][192]
// Block = 192 threads (6 waves), one token per block.
// ---------------------------------------------------------------------------
__device__ __forceinline__ float block_reduce_sum_192(float v, float* red) {
  int t = threadIdx.x;
  red[t] = v;
  if (t < 64) red[192 + t] = 0.f;   // pad to 256 for power-of-two tree
  __syncthreads();
  for (int s = 128; s > 0; s >>= 1) {
    if (t < s) red[t] += red[t + s];
    __syncthreads();
  }
  float r = red[0];
  __syncthreads();
  return r;
}

__global__ void __launch_bounds__(192) norm_rope_kernel(
    const float* __restrict__ qkv, const float* __restrict__ rope,
    const float* __restrict__ qg, const float* __restrict__ kg,
    const float* __restrict__ vg,
    _Float16* __restrict__ qh, _Float16* __restrict__ kh,
    _Float16* __restrict__ vh) {
  __shared__ float red[256];
  __shared__ float vec[256];
  int n = blockIdx.x;
  int t = threadIdx.x;
  const float* row = qkv + (size_t)n * NQKV;

  float cs = 0.f, sn = 0.f;
  if (t < DQK) { float r = rope[n * DQK + t]; cs = cosf(r); sn = sinf(r); }

  // q heads: LN (biased var, eps=1e-5) * gamma * SCALE(0.125), then rotary.
  for (int h = 0; h < H_HEADS; ++h) {
    float x = (t < DQK) ? row[h * DQK + t] : 0.f;
    float mean = block_reduce_sum_192(x, red) * (1.f / DQK);
    float d = (t < DQK) ? (x - mean) : 0.f;
    float var = block_reduce_sum_192(d * d, red) * (1.f / DQK);
    float y = d * rsqrtf(var + 1e-5f);
    if (t < DQK) vec[t] = y * qg[t] * 0.125f;
    __syncthreads();
    if (t < DQK) {
      float rh = (t < 64) ? -vec[t + 64] : vec[t - 64];
      qh[((size_t)h * NSEQ + n) * DQK + t] = (_Float16)(vec[t] * cs + rh * sn);
    }
    __syncthreads();
  }
  // k (shared across heads)
  {
    float x = (t < DQK) ? row[H_HEADS * DQK + t] : 0.f;
    float mean = block_reduce_sum_192(x, red) * (1.f / DQK);
    float d = (t < DQK) ? (x - mean) : 0.f;
    float var = block_reduce_sum_192(d * d, red) * (1.f / DQK);
    float y = d * rsqrtf(var + 1e-5f);
    if (t < DQK) vec[t] = y * kg[t];
    __syncthreads();
    if (t < DQK) {
      float rh = (t < 64) ? -vec[t + 64] : vec[t - 64];
      kh[(size_t)n * DQK + t] = (_Float16)(vec[t] * cs + rh * sn);
    }
    __syncthreads();
  }
  // v
  {
    float x = row[H_HEADS * DQK + DQK + t];
    float mean = block_reduce_sum_192(x, red) * (1.f / DV);
    float d = x - mean;
    float var = block_reduce_sum_192(d * d, red) * (1.f / DV);
    float y = d * rsqrtf(var + 1e-5f);
    vh[(size_t)n * DV + t] = (_Float16)(y * vg[t]);
  }
}

// ---------------------------------------------------------------------------
// bias[h][i][j] = sum_d gelu(rms(pw[i][j][:])) * w_bias[d][h]
// THE bandwidth-bound pass: 768 MB streamed exactly once. The stream is 4x
// the 192 MB L2 and never re-read -> non-temporal loads keep L2 for the
// operands that ARE reused (bias table, q/k/v, weights).
// Block = 8 waves; one (i,j) pair per wave; 24 floats/lane kept in registers.
// ---------------------------------------------------------------------------
__global__ void __launch_bounds__(256) bias_kernel(
    const float* __restrict__ pw, const float* __restrict__ wb,
    const float* __restrict__ rg, float* __restrict__ bias) {
  __shared__ float s_wb[DIM * H_HEADS];   // 24 KB
  __shared__ float s_g[DIM];              // 3 KB
  int t = threadIdx.x;
  for (int i = t; i < DIM * H_HEADS; i += 256) s_wb[i] = wb[i];
  for (int i = t; i < DIM; i += 256) s_g[i] = rg[i];
  __syncthreads();

  int wave = t >> 5, lane = t & 31;
  size_t pair = (size_t)blockIdx.x * 8 + wave;    // pair = i*512 + j
  const float* src = pw + pair * DIM;

  // hint the streaming pattern to L2
  __builtin_prefetch(src + 8 * DIM, 0, 0);

  f32x4 vals[6];
  float ssq = 0.f;
#pragma unroll
  for (int it = 0; it < 6; ++it) {                // lane*16B, 512B/wave/iter
    vals[it] = __builtin_nontemporal_load(((const f32x4*)src) + it * 32 + lane);
    ssq += vals[it].x * vals[it].x + vals[it].y * vals[it].y
         + vals[it].z * vals[it].z + vals[it].w * vals[it].w;
  }
#pragma unroll
  for (int m = 16; m; m >>= 1) ssq += __shfl_xor(ssq, m, 32);
  float rinv = rsqrtf(ssq * (1.f / DIM) + 1.1920929e-07f);  // torch eps=None

  float acc[8] = {0,0,0,0,0,0,0,0};
#pragma unroll
  for (int it = 0; it < 6; ++it) {
    float e4[4] = {vals[it].x, vals[it].y, vals[it].z, vals[it].w};
    int d0 = it * 128 + lane * 4;
#pragma unroll
    for (int e = 0; e < 4; ++e) {
      int d = d0 + e;
      float xn = e4[e] * rinv * s_g[d];
      float gl = 0.5f * xn * (1.f + erff(xn * 0.70710678118f));  // exact gelu
#pragma unroll
      for (int h = 0; h < 8; ++h) acc[h] += gl * s_wb[d * 8 + h];
    }
  }
#pragma unroll
  for (int h = 0; h < 8; ++h) {
#pragma unroll
    for (int m = 16; m; m >>= 1) acc[h] += __shfl_xor(acc[h], m, 32);
  }
  if (lane == 0) {
#pragma unroll
    for (int h = 0; h < 8; ++h)
      bias[(size_t)h * (NSEQ * NSEQ) + pair] = acc[h];
  }
}

// ---------------------------------------------------------------------------
// Fused attention: one block per (head, 16-row query tile), 4 waves.
//   S = QK^T + bias -> 5*tanh(S/5) -> softmax -> O = P*V  (all WMMA)
// Q tile staged to LDS via the gfx1250 async-to-LDS path (ASYNCcnt-tracked).
// ---------------------------------------------------------------------------
__global__ void __launch_bounds__(128) attn_kernel(
    const _Float16* __restrict__ qh, const _Float16* __restrict__ kh,
    const _Float16* __restrict__ vh, const float* __restrict__ bias,
    _Float16* __restrict__ O) {
  __shared__ _Float16 s_q[16 * DQK];       // 4 KB
  __shared__ float    s_sim[16 * NSEQ];    // 32 KB
  __shared__ _Float16 s_p[16 * NSEQ];      // 16 KB
  __shared__ float    s_red[16 * 8];
  __shared__ float    s_row[16];

  int h  = blockIdx.x & 7;
  int i0 = (blockIdx.x >> 3) * 16;
  int t = threadIdx.x;
  int wave = t >> 5, lane = t & 31;

  // ---- async copy of the 4 KB Q tile: 256 x b128, 2 per thread.
  // GLOBAL_LOAD_ASYNC_TO_LDS_B128: per-lane LDS dest VGPR + 64-bit global addr,
  // completion tracked with ASYNCcnt (CDNA5 ISA ch.10 / §15.18.3).
  const _Float16* qsrc = qh + ((size_t)h * NSEQ + i0) * DQK;
  unsigned lds_q = (unsigned)(size_t)(void*)s_q;   // LDS byte offset (addr[31:0])
#pragma unroll
  for (int it = 0; it < 2; ++it) {
    int e = (t + it * 128) * 8;                    // element (half) index
    unsigned dst = lds_q + (unsigned)(e * 2);      // byte address in LDS
    const _Float16* ga = qsrc + e;
    asm volatile("global_load_async_to_lds_b128 %0, %1, off"
                 :: "v"(dst), "v"(ga) : "memory");
  }
  asm volatile("s_wait_asynccnt 0x0" ::: "memory");
  __syncthreads();

  // ---- scores: wave w owns j-tiles {w, w+4, ...} (wave-uniform; EXEC full)
  const float* bsrc = bias + (size_t)h * NSEQ * NSEQ + (size_t)i0 * NSEQ;
  for (int jt = wave; jt < NSEQ / 16; jt += 4) {
    int j0 = jt * 16;
    v8f acc = {0,0,0,0,0,0,0,0};
#pragma unroll
    for (int kt = 0; kt < DQK / 32; ++kt) {
      v16h a = frag_a_rowmajor(s_q + kt * 32, DQK);
      v16h b = frag_b_transposed(kh + (size_t)j0 * DQK + kt * 32, DQK);
      acc = wmma16x16x32(a, b, acc);
    }
    int nn = lane & 15, half = lane >> 4;
#pragma unroll
    for (int r = 0; r < 8; ++r) {
      int m = half * 8 + r;
      float s = acc[r] + bsrc[(size_t)m * NSEQ + j0 + nn];
      s_sim[m * NSEQ + j0 + nn] = 5.0f * tanhf(s * 0.2f);   // softclamp
    }
  }
  __syncthreads();

  // ---- softmax over j: 8 threads per row
  int row = t >> 3, sub = t & 7;
  float mx = -1e30f;
  for (int j = sub * 64; j < sub * 64 + 64; ++j) mx = fmaxf(mx, s_sim[row * NSEQ + j]);
  s_red[row * 8 + sub] = mx;
  __syncthreads();
  if (sub == 0) {
    float m2 = s_red[row * 8];
    for (int u = 1; u < 8; ++u) m2 = fmaxf(m2, s_red[row * 8 + u]);
    s_row[row] = m2;
  }
  __syncthreads();
  float rm = s_row[row];
  float sum = 0.f;
  for (int j = sub * 64; j < sub * 64 + 64; ++j) {
    float e = expf(s_sim[row * NSEQ + j] - rm);
    s_sim[row * NSEQ + j] = e;
    sum += e;
  }
  s_red[row * 8 + sub] = sum;
  __syncthreads();
  if (sub == 0) {
    float s2 = 0.f;
    for (int u = 0; u < 8; ++u) s2 += s_red[row * 8 + u];
    s_row[row] = 1.f / s2;
  }
  __syncthreads();
  float inv = s_row[row];
  for (int j = sub * 64; j < sub * 64 + 64; ++j)
    s_p[row * NSEQ + j] = (_Float16)(s_sim[row * NSEQ + j] * inv);
  __syncthreads();

  // ---- O = P * V : wave w owns n-tiles {w, w+4, w+8} of 12
  for (int nt = wave; nt < DV / 16; nt += 4) {
    int n0 = nt * 16;
    v8f acc = {0,0,0,0,0,0,0,0};
#pragma unroll
    for (int kt = 0; kt < NSEQ / 32; ++kt) {
      v16h a = frag_a_rowmajor(s_p + kt * 32, NSEQ);
      v16h b = frag_b_rowmajor(vh + (size_t)(kt * 32) * DV + n0, DV);
      acc = wmma16x16x32(a, b, acc);
    }
    int nn = lane & 15, half = lane >> 4;
#pragma unroll
    for (int r = 0; r < 8; ++r) {
      int m = half * 8 + r;
      O[(size_t)(i0 + m) * HDV + h * DV + n0 + nn] = (_Float16)acc[r];
    }
  }
}

// ---------------------------------------------------------------------------
// Host-side orchestration
// ---------------------------------------------------------------------------
extern "C" void kernel_launch(void* const* d_in, const int* in_sizes, int n_in,
                              void* d_out, int out_size, void* d_ws, size_t ws_size,
                              hipStream_t stream) {
  (void)in_sizes; (void)n_in; (void)out_size; (void)ws_size;
  const float* x      = (const float*)d_in[0];
  const float* pw     = (const float*)d_in[1];
  const float* rope   = (const float*)d_in[2];
  const float* w_qkv  = (const float*)d_in[3];
  const float* w_out  = (const float*)d_in[4];
  const float* w_bias = (const float*)d_in[5];
  const float* qg     = (const float*)d_in[6];
  const float* kg     = (const float*)d_in[7];
  const float* vg     = (const float*)d_in[8];
  const float* rg     = (const float*)d_in[9];
  float* out = (float*)d_out;

  char* p = (char*)d_ws;
  auto carve = [&](size_t bytes) -> void* {
    void* r = (void*)p;
    p += (bytes + 255) & ~(size_t)255;
    return r;
  };
  float*    qkv    = (float*)   carve((size_t)NSEQ * NQKV * 4);         // 2.75 MB
  _Float16* x16    = (_Float16*)carve((size_t)NSEQ * DIM * 2);          // 0.75 MB
  _Float16* wqkv16 = (_Float16*)carve((size_t)DIM * NQKV * 2);          // 2.0 MB
  _Float16* wout16 = (_Float16*)carve((size_t)HDV * DIM * 2);           // 2.25 MB
  _Float16* q16    = (_Float16*)carve((size_t)H_HEADS * NSEQ * DQK * 2);// 1.0 MB
  _Float16* k16    = (_Float16*)carve((size_t)NSEQ * DQK * 2);          // 128 KB
  _Float16* v16p   = (_Float16*)carve((size_t)NSEQ * DV * 2);           // 192 KB
  float*    biasws = (float*)   carve((size_t)H_HEADS * NSEQ * NSEQ * 4);// 8 MB
  _Float16* O16    = (_Float16*)carve((size_t)NSEQ * HDV * 2);          // 1.5 MB

  // 1) fp32 -> f16 operand staging
  cvt_f32_f16<<<(NSEQ * DIM + 255) / 256, 256, 0, stream>>>(x, x16, NSEQ * DIM);
  cvt_f32_f16<<<(DIM * NQKV + 255) / 256, 256, 0, stream>>>(w_qkv, wqkv16, DIM * NQKV);
  cvt_f32_f16<<<(HDV * DIM + 255) / 256, 256, 0, stream>>>(w_out, wout16, HDV * DIM);

  // 2) qkv = x @ w_qkv    (512 x 1344 x 768, WMMA)
  gemm_f16_wmma<<<dim3(NSEQ / 16, (NQKV / 16 + 3) / 4), 128, 0, stream>>>(
      x16, wqkv16, qkv, NSEQ, NQKV, DIM);

  // 3) LayerNorms + rotary -> f16 q/k/v
  norm_rope_kernel<<<NSEQ, 192, 0, stream>>>(qkv, rope, qg, kg, vg, q16, k16, v16p);

  // 4) bias pass (the 768 MB non-temporal stream — HBM-bound critical path)
  bias_kernel<<<(NSEQ * NSEQ) / 8, 256, 0, stream>>>(pw, w_bias, rg, biasws);

  // 5) fused attention (async Q staging + WMMA scores + WMMA P*V)
  attn_kernel<<<H_HEADS * (NSEQ / 16), 128, 0, stream>>>(q16, k16, v16p, biasws, O16);

  // 6) final projection: out = O @ w_out   (512 x 768 x 1536, WMMA)
  gemm_f16_wmma<<<dim3(NSEQ / 16, (DIM / 16) / 4), 128, 0, stream>>>(
      O16, wout16, out, NSEQ, DIM, HDV);
}
